// Bert4EtWithContext_40647570489943
// MI455X (gfx1250) — matrix operand order repaired and checked by reading, not compiled, not run on hardware
//
#include <hip/hip_runtime.h>
#include <hip/hip_bf16.h>

// Problem constants
constexpr int Bn = 256;
constexpr int Ln = 512;
constexpr int Dn = 768;
constexpr int Cn = 10331;
constexpr int Kn = 3 * Dn;          // 2304
constexpr int NT = (Cn + 15) / 16;  // 646 n-tiles
constexpr int MT = Bn / 16;         // 16  m-tiles

typedef __attribute__((ext_vector_type(16))) __bf16 v16bf;
typedef __attribute__((ext_vector_type(8)))  float  v8f;

union Frag {
    unsigned d[8];
    uint4    q[2];
    v16bf    v;
};

// f32 pair -> packed bf16 dword (RNE). Prefer the hardware packed convert.
__device__ __forceinline__ unsigned pack2_bf16(float a, float b) {
#if __has_builtin(__builtin_amdgcn_cvt_pk_bf16_f32)
    auto r = __builtin_amdgcn_cvt_pk_bf16_f32(a, b);
    return __builtin_bit_cast(unsigned, r);
#else
    unsigned ua = __builtin_bit_cast(unsigned, a);
    unsigned ub = __builtin_bit_cast(unsigned, b);
    ua = (ua + 0x7FFFu + ((ua >> 16) & 1u)) >> 16;
    ub = (ub + 0x7FFFu + ((ub >> 16) & 1u)) >> 16;
    return (ua & 0xFFFFu) | (ub << 16);
#endif
}

__device__ __forceinline__ unsigned short f32_to_bf16(float a) {
    unsigned ua = __builtin_bit_cast(unsigned, a);
    ua = (ua + 0x7FFFu + ((ua >> 16) & 1u)) >> 16;
    return (unsigned short)ua;
}

// ---------------------------------------------------------------------------
// Kernel 1: gather + three masked means -> feats (B, 3D) stored as bf16 in ws
// One block (256 threads) per batch; each thread owns 3 feature dims.
// ---------------------------------------------------------------------------
__global__ __launch_bounds__(256) void pool_kernel(
    const float* __restrict__ seq,          // (B, L, D)
    const int*   __restrict__ head_index,   // (B, L)
    const int*   __restrict__ start,        // (B,)
    const int*   __restrict__ end,          // (B,)
    unsigned short* __restrict__ feats)     // (B, 3D) bf16
{
    const int b = blockIdx.x;
    const int t = threadIdx.x;

    __shared__ int s_cnt;
    if (t == 0) s_cnt = 0;
    __syncthreads();

    int local = 0;
    for (int p = t; p < Ln; p += 256)
        local += (head_index[b * Ln + p] != 0) ? 1 : 0;
    atomicAdd(&s_cnt, local);
    __syncthreads();

    const int right_len = s_cnt;
    const int s = start[b];
    const int e = end[b];

    int pmax = (e > right_len) ? e : right_len;
    if (pmax > Ln) pmax = Ln;

    float aL[3] = {0.f, 0.f, 0.f};
    float aM[3] = {0.f, 0.f, 0.f};
    float aR[3] = {0.f, 0.f, 0.f};

    const int* hi = head_index + b * Ln;
    const float* sbase = seq + (size_t)b * Ln * Dn;

    for (int p = 0; p < pmax; ++p) {
        const int idx = hi[p];
        const float* row = sbase + (size_t)idx * Dn;
        const bool mL = (p < s);
        const bool mM = (p >= s) && (p < e);
        const bool mR = (p >= e) && (p < right_len);
#pragma unroll
        for (int j = 0; j < 3; ++j) {
            const float v = row[t + j * 256];
            if (mL) aL[j] += v;
            if (mM) aM[j] += v;
            if (mR) aR[j] += v;
        }
    }

    const int sC = (s < Ln) ? s : Ln;
    const int eC = (e < Ln) ? e : Ln;
    const float cL = (float)sC;
    const float cM = (float)((eC > s) ? (eC - s) : 0);
    const float cR = (float)((right_len > e) ? (right_len - e) : 0);

    unsigned short* fb = feats + (size_t)b * Kn;
#pragma unroll
    for (int j = 0; j < 3; ++j) {
        const int d = t + j * 256;
        fb[d]          = f32_to_bf16(aL[j] / cL);
        fb[Dn + d]     = f32_to_bf16(aM[j] / cM);
        fb[2 * Dn + d] = f32_to_bf16(aR[j] / cR);
    }
}

// ---------------------------------------------------------------------------
// Kernel 2: W (C, K) f32 -> bf16 row-major in workspace. Pure bandwidth pass.
// Each thread converts 8 contiguous elements (32B in, 16B out).
// ---------------------------------------------------------------------------
__global__ __launch_bounds__(256) void wconv_kernel(
    const float* __restrict__ W,
    unsigned short* __restrict__ Wb)
{
    const size_t n = (size_t)Cn * Kn;  // 23,802,624 (multiple of 8)
    const size_t i = ((size_t)blockIdx.x * 256 + threadIdx.x) * 8;
    if (i >= n) return;

    const float4 a = *(const float4*)(W + i);
    const float4 b = *(const float4*)(W + i + 4);
    uint4 o;
    o.x = pack2_bf16(a.x, a.y);
    o.y = pack2_bf16(a.z, a.w);
    o.z = pack2_bf16(b.x, b.y);
    o.w = pack2_bf16(b.z, b.w);
    *(uint4*)(Wb + i) = o;
}

// ---------------------------------------------------------------------------
// Kernel 3a (fast path): out = feats @ Wb^T + bias, all-bf16 operands.
// WMMA bf16 16x16x32, f32 accumulate. One 16x16 tile per wave; tiles n-major
// so the 8 waves of a block share the same 16 W rows (L0/L2 reuse).
// Inner loop: 4x global_load_b128 + 1x v_wmma. No per-element VALU.
// ---------------------------------------------------------------------------
__global__ __launch_bounds__(256) void gemm_bf16_kernel(
    const unsigned short* __restrict__ feats,  // (B, K) bf16
    const unsigned short* __restrict__ Wb,     // (C, K) bf16
    const float* __restrict__ bias,
    float* __restrict__ out)
{
    const int wave = threadIdx.x >> 5;
    const int lane = threadIdx.x & 31;
    const int tileId = blockIdx.x * 8 + wave;

    const int ntile = tileId / MT;
    const int mtile = tileId % MT;

    const int hf  = lane >> 4;
    const int r16 = lane & 15;

    const int rowA = mtile * 16 + r16;
    int rowB = ntile * 16 + r16;
    if (rowB >= Cn) rowB = Cn - 1;

    // A (16-bit 16x32): lanes 0-15 hold K 0..15, lanes 16-31 hold K 16..31
    const unsigned short* Ap = feats + (size_t)rowA * Kn + hf * 16;
    // B (16-bit 32x16): lanes 0-15 hold K 0..15, lanes 16-31 hold K 16..31
    const unsigned short* Bp = Wb + (size_t)rowB * Kn + hf * 16;

    v8f acc = {};

#pragma unroll 4
    for (int k = 0; k < Kn; k += 32) {
        Frag a, b;
        a.q[0] = *(const uint4*)(Ap + k);
        a.q[1] = *(const uint4*)(Ap + k + 8);
        b.q[0] = *(const uint4*)(Bp + k);
        b.q[1] = *(const uint4*)(Bp + k + 8);

        acc = __builtin_amdgcn_wmma_f32_16x16x32_bf16(
            false, a.v, false, b.v, (short)0, acc, false, false);
    }

    const int n = ntile * 16 + r16;
    if (n < Cn) {
        const float bv = bias[n];
        float* op = out + (size_t)(mtile * 16 + hf * 8) * Cn + n;
#pragma unroll
        for (int r = 0; r < 8; ++r) op[(size_t)r * Cn] = acc[r] + bv;
    }
}

// ---------------------------------------------------------------------------
// Kernel 3b (fallback if workspace too small for bf16 W): convert W on the
// fly with packed cvt. Same tiling.
// ---------------------------------------------------------------------------
__global__ __launch_bounds__(256) void gemm_otf_kernel(
    const unsigned short* __restrict__ feats,  // (B, K) bf16
    const float* __restrict__ W,               // (C, K) f32
    const float* __restrict__ bias,
    float* __restrict__ out)
{
    const int wave = threadIdx.x >> 5;
    const int lane = threadIdx.x & 31;
    const int tileId = blockIdx.x * 8 + wave;

    const int ntile = tileId / MT;
    const int mtile = tileId % MT;

    const int hf  = lane >> 4;
    const int r16 = lane & 15;

    const int rowA = mtile * 16 + r16;
    int rowB = ntile * 16 + r16;
    if (rowB >= Cn) rowB = Cn - 1;

    const unsigned short* Ap = feats + (size_t)rowA * Kn + hf * 16;
    const float* Bp = W + (size_t)rowB * Kn + hf * 16;

    v8f acc = {};

#pragma unroll 2
    for (int k = 0; k < Kn; k += 32) {
        Frag a;
        a.q[0] = *(const uint4*)(Ap + k);
        a.q[1] = *(const uint4*)(Ap + k + 8);

        const float4* bp = (const float4*)(Bp + k);
        const float4 w0 = bp[0];
        const float4 w1 = bp[1];
        const float4 w2 = bp[2];
        const float4 w3 = bp[3];

        Frag b;
        b.d[0] = pack2_bf16(w0.x, w0.y);
        b.d[1] = pack2_bf16(w0.z, w0.w);
        b.d[2] = pack2_bf16(w1.x, w1.y);
        b.d[3] = pack2_bf16(w1.z, w1.w);
        b.d[4] = pack2_bf16(w2.x, w2.y);
        b.d[5] = pack2_bf16(w2.z, w2.w);
        b.d[6] = pack2_bf16(w3.x, w3.y);
        b.d[7] = pack2_bf16(w3.z, w3.w);

        acc = __builtin_amdgcn_wmma_f32_16x16x32_bf16(
            false, a.v, false, b.v, (short)0, acc, false, false);
    }

    const int n = ntile * 16 + r16;
    if (n < Cn) {
        const float bv = bias[n];
        float* op = out + (size_t)(mtile * 16 + hf * 8) * Cn + n;
#pragma unroll
        for (int r = 0; r < 8; ++r) op[(size_t)r * Cn] = acc[r] + bv;
    }
}

extern "C" void kernel_launch(void* const* d_in, const int* in_sizes, int n_in,
                              void* d_out, int out_size, void* d_ws, size_t ws_size,
                              hipStream_t stream) {
    const float* seq        = (const float*)d_in[0];
    const int*   head_index = (const int*)  d_in[1];
    const int*   start      = (const int*)  d_in[2];
    const int*   end        = (const int*)  d_in[3];
    const float* W          = (const float*)d_in[4];
    const float* bias       = (const float*)d_in[5];
    float* out = (float*)d_out;

    const size_t featsBytes = (size_t)Bn * Kn * sizeof(unsigned short); // 1.18 MB
    const size_t wBytes     = (size_t)Cn * Kn * sizeof(unsigned short); // 47.6 MB

    unsigned short* feats = (unsigned short*)d_ws;

    pool_kernel<<<Bn, 256, 0, stream>>>(seq, head_index, start, end, feats);

    const int totalTiles = NT * MT;        // 10336
    const int gemmBlocks = totalTiles / 8; // 1292

    if (ws_size >= featsBytes + wBytes) {
        // Fast path: one bandwidth-bound conversion pass, then a pure
        // load_b128 + wmma inner loop (bf16 W stays L2-resident, 192 MB L2).
        unsigned short* Wb = feats + (size_t)Bn * Kn;
        const size_t nElem = (size_t)Cn * Kn;
        const int convBlocks = (int)((nElem / 8 + 255) / 256);
        wconv_kernel<<<convBlocks, 256, 0, stream>>>(W, Wb);
        gemm_bf16_kernel<<<gemmBlocks, 256, 0, stream>>>(feats, Wb, bias, out);
    } else {
        gemm_otf_kernel<<<gemmBlocks, 256, 0, stream>>>(feats, W, bias, out);
    }
}
